// DeepKMeans_63814624084665
// MI455X (gfx1250) — matrix operand order repaired and checked by reading, not compiled, not run on hardware
//
#include <hip/hip_runtime.h>
#include <hip/hip_bf16.h>

#define ALPHA_C 1000.0f

typedef __attribute__((ext_vector_type(16))) __bf16 v16bf;
typedef __attribute__((ext_vector_type(8)))  float  v8f;
typedef __attribute__((ext_vector_type(2)))  float  v2f;
typedef __attribute__((ext_vector_type(4)))  int    v4i;

#define GLOBAL_AS __attribute__((address_space(1)))
#define LDS_AS    __attribute__((address_space(3)))

#if defined(__gfx1250__) && __has_builtin(__builtin_amdgcn_global_load_async_to_lds_b128)
#define HAVE_ASYNC_LDS 1
#else
#define HAVE_ASYNC_LDS 0
#endif

static constexpr int Bn = 1024;  // batch
static constexpr int Dn = 4096;  // d_in
static constexpr int En = 512;   // embed dim == #clusters

// ---------------------------------------------------------------------------
// f32 -> bf16 conversion (grid-stride)
// ---------------------------------------------------------------------------
__global__ void convert_bf16_kernel(const float* __restrict__ src,
                                    __bf16* __restrict__ dst, int n) {
  int i = blockIdx.x * blockDim.x + threadIdx.x;
  int stride = gridDim.x * blockDim.x;
  for (; i < n; i += stride) dst[i] = (__bf16)src[i];
}

// ---------------------------------------------------------------------------
// bf16 GEMM: C[M,N] = A[M,K] @ B[K,N] (+ bias[N]), f32 accumulate via
// v_wmma_f32_16x16x32_bf16.
//   Block tile 128x64, 8 waves (4x2), 32x32 per wave = 2x2 WMMA register
//   block (A/B fragment reuse halves LDS traffic per WMMA). K-step 32.
//   A staged global->LDS with GLOBAL_LOAD_ASYNC_TO_LDS_B128 (ASYNCcnt path);
//   B transposed while staging through VGPRs.
// MODE 0: write f32 C and bf16 C (embedding).  MODE 1: f32 C only.
// ---------------------------------------------------------------------------
template <int MODE>
__global__ __launch_bounds__(256)
void gemm_bf16_kernel(const __bf16* __restrict__ A, const __bf16* __restrict__ Bm,
                      const float* __restrict__ bias,
                      float* __restrict__ Cf, __bf16* __restrict__ Cbf,
                      int M, int N, int K) {
  // 80B row stride: every 16B async chunk lands aligned; rows hit distinct banks
  __shared__ __align__(16) unsigned short As[128][40];  // [m][k]
  __shared__ unsigned short Bt[64][34];                 // [n][k] (transposed)

  const int tid  = threadIdx.x;
  const int lane = tid & 31;
  const int wave = tid >> 5;
  const int wm   = (wave & 3) * 32;   // wave M offset in block tile
  const int wn   = (wave >> 2) * 32;  // wave N offset in block tile
  const int m0   = blockIdx.x * 128;
  const int n0   = blockIdx.y * 64;

  const int aRow = tid >> 2;          // 0..63 (+64 for second half)
  const int aCol = (tid & 3) * 8;     // 16B chunk within 32-wide K slab
  const int bK   = tid >> 3;          // 0..31
  const int bN   = (tid & 7) * 8;     // 0..56

  v8f acc[2][2] = {};

  for (int k0 = 0; k0 < K; k0 += 32) {
    // ---- stage A (128x32 bf16) ----
#if HAVE_ASYNC_LDS
#pragma unroll
    for (int h = 0; h < 2; ++h) {
      const int r = aRow + h * 64;
      __builtin_amdgcn_global_load_async_to_lds_b128(
          (GLOBAL_AS v4i*)(A + (size_t)(m0 + r) * K + k0 + aCol),
          (LDS_AS v4i*)&As[r][aCol], 0, 0);
    }
#else
#pragma unroll
    for (int h = 0; h < 2; ++h) {
      const int r = aRow + h * 64;
      const __bf16* ag = A + (size_t)(m0 + r) * K + k0 + aCol;
#pragma unroll
      for (int j = 0; j < 8; ++j)
        As[r][aCol + j] = __builtin_bit_cast(unsigned short, ag[j]);
    }
#endif

    // ---- stage B transposed (32k x 64n -> Bt[n][k]) ----
    const __bf16* bg = Bm + (size_t)(k0 + bK) * N + n0 + bN;
    __builtin_prefetch(bg + (size_t)32 * N, 0, 1);  // next K-slab
#pragma unroll
    for (int j = 0; j < 8; ++j)
      Bt[bN + j][bK] = __builtin_bit_cast(unsigned short, bg[j]);

#if HAVE_ASYNC_LDS
    asm volatile("s_wait_asynccnt 0" ::: "memory");
#endif
    __syncthreads();

    // ---- fragments (documented 16-bit A/B VGPR layouts) ----
    v16bf af[2], bfr[2];
    const int kb  = (lane >> 4) * 8;
    const int kb2 = (lane >> 4) * 16;
#pragma unroll
    for (int t = 0; t < 2; ++t) {
      const int ar = wm + t * 16 + (lane & 15);
#pragma unroll
      for (int e = 0; e < 8; ++e) {
        af[t][e]     = __builtin_bit_cast(__bf16, As[ar][kb + e]);
        af[t][e + 8] = __builtin_bit_cast(__bf16, As[ar][16 + kb + e]);
      }
      const int bc = wn + t * 16 + (lane & 15);
#pragma unroll
      for (int e = 0; e < 8; ++e) {
        bfr[t][e]     = __builtin_bit_cast(__bf16, Bt[bc][kb + e]);
        bfr[t][e + 8] = __builtin_bit_cast(__bf16, Bt[bc][16 + kb + e]);
      }
    }

    // ---- 2x2 register block: 4 WMMAs per fragment set ----
#pragma unroll
    for (int mt = 0; mt < 2; ++mt)
#pragma unroll
      for (int nt = 0; nt < 2; ++nt)
        acc[mt][nt] = __builtin_amdgcn_wmma_f32_16x16x32_bf16(
            false, af[mt], false, bfr[nt], (short)0, acc[mt][nt], false, false);

    __syncthreads();
  }

  // ---- epilogue: C VGPR r -> (lanes 0-15: M=+r) (lanes 16-31: M=+8+r) ----
#pragma unroll
  for (int mt = 0; mt < 2; ++mt) {
#pragma unroll
    for (int nt = 0; nt < 2; ++nt) {
      const int col = n0 + wn + nt * 16 + (lane & 15);
      const float bv = bias ? bias[col] : 0.0f;
#pragma unroll
      for (int r = 0; r < 8; ++r) {
        const int row = m0 + wm + mt * 16 + r + (lane >> 4) * 8;
        const float v = acc[mt][nt][r] + bv;
        Cf[(size_t)row * N + col] = v;
        if (MODE == 0) Cbf[(size_t)row * N + col] = (__bf16)v;
      }
    }
  }
}

// ---------------------------------------------------------------------------
// f32 distance GEMM: dot[b,k] = emb[b,:] . creps[k,:] via v_wmma_f32_16x16x4_f32
// dist = ||e||^2 + ||c||^2 - 2*dot fused in the epilogue (f32 kept: the
// exp(-1000*dd) epilogue is effectively an argmin and is precision-critical).
// ---------------------------------------------------------------------------
__global__ __launch_bounds__(256)
void dist_wmma_kernel(const float* __restrict__ emb, const float* __restrict__ creps,
                      const float* __restrict__ enorm, const float* __restrict__ cnorm,
                      float* __restrict__ dist) {
  __shared__ float Asf[64][17];   // [b][e]
  __shared__ float Btf[32][17];   // [kcluster][e]

  const int tid  = threadIdx.x;
  const int lane = tid & 31;
  const int wave = tid >> 5;
  const int wm   = (wave & 3) * 16;
  const int wn   = (wave >> 2) * 16;
  const int m0   = blockIdx.x * 64;
  const int n0   = blockIdx.y * 32;

  const int aRow = tid >> 2;        // 0..63
  const int aCol = (tid & 3) * 4;   // 0..12
  const int bK   = tid >> 3;        // 0..31  (cluster row)
  const int bE   = (tid & 7) * 2;   // 0..14

  v8f acc = {};

  for (int e0 = 0; e0 < En; e0 += 16) {
#pragma unroll
    for (int j = 0; j < 4; ++j)
      Asf[aRow][aCol + j] = emb[(size_t)(m0 + aRow) * En + e0 + aCol + j];
#pragma unroll
    for (int j = 0; j < 2; ++j)
      Btf[bK][bE + j] = creps[(size_t)(n0 + bK) * En + e0 + bE + j];
    __syncthreads();

    const int ar = wm + (lane & 15);
    const int bc = wn + (lane & 15);
#pragma unroll
    for (int kk = 0; kk < 16; kk += 4) {
      const int ka = kk + (lane >> 4) * 2;
      v2f a2, b2;
      a2[0] = Asf[ar][ka];
      a2[1] = Asf[ar][ka + 1];
      b2[0] = Btf[bc][ka];
      b2[1] = Btf[bc][ka + 1];
      acc = __builtin_amdgcn_wmma_f32_16x16x4_f32(
          false, a2, false, b2, (short)0, acc, false, false);
    }
    __syncthreads();
  }

  const int col = n0 + wn + (lane & 15);
  const float cn = cnorm[col];
#pragma unroll
  for (int r = 0; r < 8; ++r) {
    const int row = m0 + wm + r + (lane >> 4) * 8;
    dist[(size_t)row * En + col] = enorm[row] + cn - 2.0f * acc[r];
  }
}

// ---------------------------------------------------------------------------
// Row squared-norms: out[r] = sum_c Mat[r][c]^2
// ---------------------------------------------------------------------------
__global__ void rownorm_kernel(const float* __restrict__ Mat,
                               float* __restrict__ out, int rows, int cols) {
  int r = blockIdx.x * blockDim.x + threadIdx.x;
  if (r >= rows) return;
  const float* p = Mat + (size_t)r * cols;
  float s = 0.0f;
  for (int c = 0; c < cols; ++c) { float v = p[c]; s += v * v; }
  out[r] = s;
}

// ---------------------------------------------------------------------------
// Row-wise shifted-exp weighting: one block (256 thr) per row of 512.
// ---------------------------------------------------------------------------
__global__ __launch_bounds__(256)
void weight_kernel(const float* __restrict__ dist, float* __restrict__ wout) {
  __shared__ float red[256];
  const int row = blockIdx.x;
  const int tid = threadIdx.x;
  const float* d = dist + (size_t)row * En;
  const float d0 = d[tid];
  const float d1 = d[tid + 256];

  red[tid] = fminf(d0, d1);
  __syncthreads();
  for (int s = 128; s > 0; s >>= 1) {
    if (tid < s) red[tid] = fminf(red[tid], red[tid + s]);
    __syncthreads();
  }
  const float mn = red[0];
  __syncthreads();

  const float e0 = __expf(-ALPHA_C * (d0 - mn));
  const float e1 = __expf(-ALPHA_C * (d1 - mn));
  red[tid] = e0 + e1;
  __syncthreads();
  for (int s = 128; s > 0; s >>= 1) {
    if (tid < s) red[tid] += red[tid + s];
    __syncthreads();
  }
  const float inv = 1.0f / red[0];

  wout[(size_t)row * En + tid]       = d0 * e0 * inv;
  wout[(size_t)row * En + tid + 256] = d1 * e1 * inv;
}

// ---------------------------------------------------------------------------
// Launch
// ---------------------------------------------------------------------------
extern "C" void kernel_launch(void* const* d_in, const int* in_sizes, int n_in,
                              void* d_out, int out_size, void* d_ws, size_t ws_size,
                              hipStream_t stream) {
  const float* x     = (const float*)d_in[0];   // [1024,4096]
  const float* creps = (const float*)d_in[1];   // [512,512]
  const float* Wenc  = (const float*)d_in[2];   // [4096,512]
  const float* benc  = (const float*)d_in[3];   // [512]
  const float* Wdec  = (const float*)d_in[4];   // [512,4096]
  const float* bdec  = (const float*)d_in[5];   // [4096]

  float* out   = (float*)d_out;
  float* wdist = out;                         // [1024,512]
  float* dist  = out + (size_t)Bn * En;       // [1024,512]
  float* recon = out + (size_t)2 * Bn * En;   // [1024,4096]

  char* ws = (char*)d_ws;
  size_t off = 0;
  auto alloc = [&](size_t bytes) -> void* {
    void* p = ws + off;
    off = (off + bytes + 255) & ~(size_t)255;
    return p;
  };
  __bf16* x_bf   = (__bf16*)alloc((size_t)Bn * Dn * sizeof(__bf16));
  __bf16* we_bf  = (__bf16*)alloc((size_t)Dn * En * sizeof(__bf16));
  __bf16* wd_bf  = (__bf16*)alloc((size_t)En * Dn * sizeof(__bf16));
  float*  emb    = (float*)alloc((size_t)Bn * En * sizeof(float));
  __bf16* emb_bf = (__bf16*)alloc((size_t)Bn * En * sizeof(__bf16));
  float*  enorm  = (float*)alloc((size_t)Bn * sizeof(float));
  float*  cnorm  = (float*)alloc((size_t)En * sizeof(float));

  // 1) precision conversion (read-once; halves GEMM streaming traffic)
  convert_bf16_kernel<<<1024, 256, 0, stream>>>(x, x_bf, Bn * Dn);
  convert_bf16_kernel<<<1024, 256, 0, stream>>>(Wenc, we_bf, Dn * En);
  convert_bf16_kernel<<<1024, 256, 0, stream>>>(Wdec, wd_bf, En * Dn);

  // 2) embedding = x @ W_enc + b_enc   (bf16 WMMA, f32 accum)
  dim3 g1(Bn / 128, En / 64);   // 8 x 8
  gemm_bf16_kernel<0><<<g1, 256, 0, stream>>>(x_bf, we_bf, benc, emb, emb_bf,
                                              Bn, En, Dn);

  // 3) reconstruction = emb @ W_dec + b_dec
  dim3 g2(Bn / 128, Dn / 64);   // 8 x 64
  gemm_bf16_kernel<1><<<g2, 256, 0, stream>>>(emb_bf, wd_bf, bdec, recon, nullptr,
                                              Bn, Dn, En);

  // 4) squared norms for the distance expansion
  rownorm_kernel<<<(Bn + 255) / 256, 256, 0, stream>>>(emb, enorm, Bn, En);
  rownorm_kernel<<<(En + 255) / 256, 256, 0, stream>>>(creps, cnorm, En, En);

  // 5) distances via f32 WMMA
  dim3 g3(Bn / 64, En / 32);
  dist_wmma_kernel<<<g3, 256, 0, stream>>>(emb, creps, enorm, cnorm, dist);

  // 6) weighted distances
  weight_kernel<<<Bn, 256, 0, stream>>>(dist, wdist);
}
